// ImprovedGAT_65524021068101
// MI455X (gfx1250) — matrix operand order repaired and checked by reading, not compiled, not run on hardware
//
#include <hip/hip_runtime.h>
#include <hip/hip_bf16.h>
#include <math.h>

// ---------------------------------------------------------------------------
// 3-layer GAT for MI455X (gfx1250, wave32).
//  - GEMMs on the WMMA pipe: bf16 A/B, f32 accumulate, fragment-ready weight
//    layout (two b128 loads per operand per WMMA, zero per-element VALU).
//  - Compile-time output strides -> immediate-offset C stores.
//  - Edge softmax/aggregate via L2-resident float atomics.
//  - Self-loops folded in analytically; softmax normalization factored out of
//    the scatter-add (2 edge passes per layer instead of 3).
// ---------------------------------------------------------------------------

#define NEG_SLOPE 0.2f
#define IN_CH 128

typedef __attribute__((ext_vector_type(16))) __bf16 v16bf;
typedef __attribute__((ext_vector_type(8)))  float  v8f;

union V16U { v16bf v; uint4 u[2]; };

static __device__ __forceinline__ v16bf mkv16(uint4 lo, uint4 hi) {
  V16U t; t.u[0] = lo; t.u[1] = hi; return t.v;
}

static __device__ __forceinline__ unsigned short f2bf_bits(float f) {
  // round-to-nearest-even fp32 -> bf16
  unsigned u = __float_as_uint(f);
  return (unsigned short)((u + 0x7FFFu + ((u >> 16) & 1u)) >> 16);
}
static __device__ __forceinline__ __bf16 f2bf(float f) {
  unsigned short s = f2bf_bits(f);
  return __builtin_bit_cast(__bf16, s);
}

static __device__ __forceinline__ float leaky(float x) {
  return x > 0.f ? x : NEG_SLOPE * x;
}
static __device__ __forceinline__ float eluf(float x) {
  return x > 0.f ? x : (__expf(x) - 1.f);
}

// sign-aware float atomic max: int-max for >=0, uint-min for <0.
static __device__ __forceinline__ void atomicMaxF(float* addr, float v) {
  if (v >= 0.f) atomicMax((int*)addr, __float_as_int(v));
  else          atomicMin((unsigned int*)addr, __float_as_uint(v));
}

// ---------------------------------------------------------------------------
// GEMM: C[nrows,HCA] = A[nrows,K] @ W[K,HCA]
//   A: bf16 row-major. Wf: bf16, fragment-ready [CT][KT][lane][16] (columns
//   padded to HCP, multiple of 16). C: f32 row-major, stride HCA (template
//   constant -> immediate-offset stores).
// Block: 256 threads = 8 waves; RT row-tiles staged in LDS; waves sweep
// (row-tile, col-tile) tasks. All fragment loads are contiguous b128.
// ---------------------------------------------------------------------------
template <int K, int HCP, int HCA, int RT>
__global__ __launch_bounds__(256)
void gemm_wmma_bf16(const __bf16* __restrict__ A, const __bf16* __restrict__ Wf,
                    float* __restrict__ Cout, int nrows) {
  constexpr int CT = HCP / 16;   // column tiles
  constexpr int KT = K / 32;     // k-step tiles
  __shared__ __bf16 sA[RT * 16 * K];

  const int blockRow0 = blockIdx.x * (RT * 16);

  // Stage RT*16 rows of A into LDS with 16B copies.
  constexpr int CHUNKS = RT * 16 * K / 8;   // 8 bf16 per 16B chunk
  constexpr int ROWCH  = K / 8;
  for (int c = threadIdx.x; c < CHUNKS; c += blockDim.x) {
    int r  = c / ROWCH;
    int wr = c - r * ROWCH;
    int gr = blockRow0 + r;
    uint4 v = make_uint4(0u, 0u, 0u, 0u);
    if (gr < nrows) v = *(const uint4*)(A + (size_t)gr * K + wr * 8);
    *(uint4*)(&sA[r * K + wr * 8]) = v;
  }
  __syncthreads();

  const int wave = threadIdx.x >> 5;
  const int lane = threadIdx.x & 31;
  const int half = lane >> 4;
  const int lr   = lane & 15;

  for (int t = wave; t < RT * CT; t += 8) {
    const int rt   = t / CT;
    const int ct   = t - rt * CT;
    const int row0 = blockRow0 + rt * 16;
    if (row0 >= nrows) continue;
    const int col  = ct * 16 + lr;
    v8f acc = {};
#pragma unroll
    for (int kt = 0; kt < KT; ++kt) {
      // A 16x32 bf16 fragment: lane lr = row; elements K = kt*32 +
      // {0..7 (+16..23)} + half*8  -> two contiguous 16B LDS loads.
      const __bf16* ap = &sA[(rt * 16 + lr) * K + kt * 32 + half * 8];
      v16bf a = mkv16(*(const uint4*)ap, *(const uint4*)(ap + 16));
      // B fragment: pre-shuffled, 32B contiguous per lane.
      const __bf16* bp = Wf + (((size_t)ct * KT + kt) * 32 + lane) * 16;
      v16bf b = mkv16(*(const uint4*)bp, *(const uint4*)(bp + 8));
      acc = __builtin_amdgcn_wmma_f32_16x16x32_bf16(
          /*neg_a=*/false, a, /*neg_b=*/false, b,
          /*c_mod=*/(short)0, acc, /*reuse_a=*/false, /*reuse_b=*/false);
    }
    // f32 C/D layout: VGPR j -> M = j + 8*half, N = lr.
    if (HCP == HCA || col < HCA) {
      float* base = Cout + (size_t)(row0 + 8 * half) * HCA + col;
      if (row0 + 16 <= nrows) {            // full tile: immediate-offset stores
#pragma unroll
        for (int j = 0; j < 8; ++j) base[j * HCA] = acc[j];
      } else {                             // tail tile (at most one block)
#pragma unroll
        for (int j = 0; j < 8; ++j)
          if (row0 + 8 * half + j < nrows) base[j * HCA] = acc[j];
      }
    }
  }
}

// Shuffle W[K,HC] f32 into fragment-ready bf16 Wf[CT][KT][lane][16], zero-pad
// columns HC..CT*16.
__global__ void weight_frag_kernel(const float* __restrict__ W,
                                   __bf16* __restrict__ Wf,
                                   int K, int HC, int CT) {
  int idx = blockIdx.x * blockDim.x + threadIdx.x;
  int KT = K / 32;
  int total = CT * KT * 512;
  if (idx >= total) return;
  int i    = idx & 15;
  int lane = (idx >> 4) & 31;
  int tile = idx >> 9;            // ct*KT + kt
  int kt   = tile % KT;
  int ct   = tile / KT;
  int half = lane >> 4, lr = lane & 15;
  int row = kt * 32 + half * 8 + i + (i >= 8 ? 8 : 0);    // B 32x16: lanes 0-15 K 0..7,16..23; 16-31 K 8..15,24..31
  int col = ct * 16 + lr;
  float v = (col < HC) ? W[(size_t)row * HC + col] : 0.f;
  Wf[idx] = f2bf(v);
}

// f32 -> bf16, 4 elements per thread, packed 8B stores.
__global__ void f32_to_bf16_kernel(const float* __restrict__ in,
                                   __bf16* __restrict__ out, int n4) {
  int i = blockIdx.x * blockDim.x + threadIdx.x;
  if (i >= n4) return;
  float4 v = ((const float4*)in)[i];
  uint2 p;
  p.x = (unsigned)f2bf_bits(v.x) | ((unsigned)f2bf_bits(v.y) << 16);
  p.y = (unsigned)f2bf_bits(v.z) | ((unsigned)f2bf_bits(v.w) << 16);
  ((uint2*)out)[i] = p;
}

// a_s[n,h] = <h[n,h,:], att_src[h,:]> ; same for a_d. One thread per (n,h).
__global__ void attn_coef_kernel(const float* __restrict__ h,
                                 const float* __restrict__ attS,
                                 const float* __restrict__ attD,
                                 float* __restrict__ a_s, float* __restrict__ a_d,
                                 int N, int H, int C) {
  int i = blockIdx.x * blockDim.x + threadIdx.x;
  if (i >= N * H) return;
  int n = i / H, hh = i - n * H;
  const float4* hp = (const float4*)(h + (size_t)n * H * C + hh * C);
  const float4* sp = (const float4*)(attS + hh * C);
  const float4* dp = (const float4*)(attD + hh * C);
  float ss = 0.f, dd = 0.f;
  for (int c = 0; c < C / 4; ++c) {
    float4 hv = hp[c], sv = sp[c], dv = dp[c];
    ss += hv.x * sv.x + hv.y * sv.y + hv.z * sv.z + hv.w * sv.w;
    dd += hv.x * dv.x + hv.y * dv.y + hv.z * dv.z + hv.w * dv.w;
  }
  a_s[i] = ss;
  a_d[i] = dd;
}

// m[n,h] initialized with the self-loop logit (no materialized self-edges).
__global__ void seg_max_init_kernel(const float* __restrict__ a_s,
                                    const float* __restrict__ a_d,
                                    float* __restrict__ m, int NH) {
  int i = blockIdx.x * blockDim.x + threadIdx.x;
  if (i >= NH) return;
  m[i] = leaky(a_s[i] + a_d[i]);
}

// Edge pass 1: segment max of leaky(a_s[src]+a_d[dst]) over dst.
__global__ void edge_max_kernel(const long long* __restrict__ edges, int E,
                                const float* __restrict__ a_s,
                                const float* __restrict__ a_d,
                                float* __restrict__ m, int H) {
  int idx = blockIdx.x * blockDim.x + threadIdx.x;
  if (idx >= E * H) return;
  int e = idx / H, hh = idx - e * H;
  long long s = edges[e];
  long long d = edges[(size_t)E + e];
  float ev = leaky(a_s[s * H + hh] + a_d[d * H + hh]);
  atomicMaxF(&m[d * H + hh], ev);
}

// Self-loop contribution: initializes denom and out (plain stores, no memset).
__global__ void self_contrib_kernel(const float* __restrict__ h,
                                    const float* __restrict__ a_s,
                                    const float* __restrict__ a_d,
                                    const float* __restrict__ m,
                                    float* __restrict__ den,
                                    float* __restrict__ out,
                                    int N, int H, int C) {
  int i = blockIdx.x * blockDim.x + threadIdx.x;
  if (i >= N * H) return;
  int n = i / H, hh = i - n * H;
  float w = __expf(leaky(a_s[i] + a_d[i]) - m[i]);
  den[i] = w;
  const float* hp = h + (size_t)n * H * C + hh * C;
  float* op = out + (size_t)n * H * C + hh * C;
  for (int c = 0; c < C; ++c) op[c] = w * hp[c];
}

// Edge pass 2: one wave (32 lanes) per edge. Unnormalized weighted scatter-add;
// normalization (divide by denom) deferred to the node epilogue.
__global__ __launch_bounds__(256)
void edge_agg_kernel(const long long* __restrict__ edges, int E,
                     const float* __restrict__ h,
                     const float* __restrict__ a_s,
                     const float* __restrict__ a_d,
                     const float* __restrict__ m,
                     float* __restrict__ den, float* __restrict__ out,
                     int H, int C) {
  int wid = (blockIdx.x * blockDim.x + threadIdx.x) >> 5;
  if (wid >= E) return;
  int lane = threadIdx.x & 31;
  long long s = edges[wid];
  long long d = edges[(size_t)E + wid];
  const float* hrow = h + (size_t)s * H * C;
  float* orow = out + (size_t)d * H * C;
  __builtin_prefetch(hrow, 0, 3);
  for (int hh = 0; hh < H; ++hh) {
    float ev = leaky(a_s[s * H + hh] + a_d[d * H + hh]);
    float w = __expf(ev - m[d * H + hh]);
    if (lane == 0) atomicAdd(&den[d * H + hh], w);
    const float* hp = hrow + hh * C;
    float* op = orow + hh * C;
    for (int c = lane; c < C; c += 32) atomicAdd(&op[c], w * hp[c]);
  }
}

// Epilogue (layers 1,2): out/denom + bias, ELU, emit bf16 features for the
// next GEMM. May alias `out` (elementwise).
__global__ void finalize_bf16_kernel(const float* __restrict__ out,
                                     const float* __restrict__ den,
                                     const float* __restrict__ bias,
                                     __bf16* __restrict__ dst,
                                     int N, int H, int C) {
  int i = blockIdx.x * blockDim.x + threadIdx.x;
  int HC = H * C;
  if (i >= N * HC) return;
  int n = i / HC, j = i - n * HC;
  float v = out[i] / den[n * H + j / C] + bias[j];
  dst[i] = f2bf(eluf(v));
}

// Epilogue (layer 3): logits in f32, no ELU. May alias `out`.
__global__ void finalize_f32_kernel(const float* __restrict__ out,
                                    const float* __restrict__ den,
                                    const float* __restrict__ bias,
                                    float* __restrict__ dst,
                                    int N, int H, int C) {
  int i = blockIdx.x * blockDim.x + threadIdx.x;
  int HC = H * C;
  if (i >= N * HC) return;
  int n = i / HC, j = i - n * HC;
  dst[i] = out[i] / den[n * H + j / C] + bias[j];
}

// One wave per node; C<=64 covered by lane and lane+32.
__global__ void log_softmax_kernel(const float* __restrict__ in,
                                   float* __restrict__ out, int N, int C) {
  int w = (blockIdx.x * blockDim.x + threadIdx.x) >> 5;
  int lane = threadIdx.x & 31;
  if (w >= N) return;
  const float* p = in + (size_t)w * C;
  float v0 = (lane < C) ? p[lane] : -INFINITY;
  float v1 = (lane + 32 < C) ? p[lane + 32] : -INFINITY;
  float mx = fmaxf(v0, v1);
  for (int o = 16; o > 0; o >>= 1) mx = fmaxf(mx, __shfl_xor(mx, o));
  float sum = ((lane < C) ? __expf(v0 - mx) : 0.f) +
              ((lane + 32 < C) ? __expf(v1 - mx) : 0.f);
  for (int o = 16; o > 0; o >>= 1) sum += __shfl_xor(sum, o);
  float lg = __logf(sum);
  if (lane < C) out[(size_t)w * C + lane] = v0 - mx - lg;
  if (lane + 32 < C) out[(size_t)w * C + lane + 32] = v1 - mx - lg;
}

// ---------------------------------------------------------------------------

extern "C" void kernel_launch(void* const* d_in, const int* in_sizes, int n_in,
                              void* d_out, int out_size, void* d_ws, size_t ws_size,
                              hipStream_t stream) {
  const float*     x  = (const float*)d_in[0];
  const long long* ei = (const long long*)d_in[1];    // int64 edge_index [2,E]
  const float* W1  = (const float*)d_in[2];
  const float* as1 = (const float*)d_in[3];
  const float* ad1 = (const float*)d_in[4];
  const float* b1  = (const float*)d_in[5];
  const float* W2  = (const float*)d_in[6];
  const float* as2 = (const float*)d_in[7];
  const float* ad2 = (const float*)d_in[8];
  const float* b2  = (const float*)d_in[9];
  const float* W3  = (const float*)d_in[10];
  const float* as3 = (const float*)d_in[11];
  const float* ad3 = (const float*)d_in[12];
  const float* b3  = (const float*)d_in[13];

  const int N = in_sizes[0] / IN_CH;
  const int E = in_sizes[1] / 2;

  // ---- workspace layout (bump allocator, 256B aligned; ~286MB @ N=100k) ----
  char* ws = (char*)d_ws;
  size_t off = 0;
  auto alloc = [&](size_t bytes) {
    size_t o = off;
    off = (off + bytes + 255) & ~(size_t)255;
    return (void*)(ws + o);
  };
  float*  hbuf  = (float*)alloc((size_t)N * 256 * sizeof(float)); // h per layer
  float*  obuf  = (float*)alloc((size_t)N * 256 * sizeof(float)); // out L1/L3
  float*  sbuf  = (float*)alloc((size_t)N * 40  * sizeof(float)); // out L2
  __bf16* featb = (__bf16*)alloc((size_t)N * 256 * sizeof(__bf16)); // xb/feat2/feat3
  float*  a_s   = (float*)alloc((size_t)N * 8 * sizeof(float));
  float*  a_d   = (float*)alloc((size_t)N * 8 * sizeof(float));
  float*  mbf   = (float*)alloc((size_t)N * 8 * sizeof(float));
  float*  den   = (float*)alloc((size_t)N * 8 * sizeof(float));
  __bf16* wf1   = (__bf16*)alloc((size_t)16 * 4 * 512 * sizeof(__bf16)); // 128->256
  __bf16* wf2   = (__bf16*)alloc((size_t)2  * 8 * 512 * sizeof(__bf16)); // 256->32
  __bf16* wf3   = (__bf16*)alloc((size_t)3  * 1 * 512 * sizeof(__bf16)); // 32->48(40)

  // ---- one-time conversions (cheap; weights are KB-scale, x is 51MB) ----
  {
    int n4 = N * IN_CH / 4;
    f32_to_bf16_kernel<<<(n4 + 255) / 256, 256, 0, stream>>>(x, featb, n4);
    weight_frag_kernel<<<(16 * 4 * 512 + 255) / 256, 256, 0, stream>>>(W1, wf1, 128, 256, 16);
    weight_frag_kernel<<<(2 * 8 * 512 + 255) / 256, 256, 0, stream>>>(W2, wf2, 256, 32, 2);
    weight_frag_kernel<<<(3 * 1 * 512 + 255) / 256, 256, 0, stream>>>(W3, wf3, 32, 40, 3);
  }

  auto edge_phase = [&](const float* h, const float* attS, const float* attD,
                        int H, int C, float* out) {
    const int NH = N * H;
    attn_coef_kernel<<<(NH + 255) / 256, 256, 0, stream>>>(h, attS, attD, a_s, a_d, N, H, C);
    seg_max_init_kernel<<<(NH + 255) / 256, 256, 0, stream>>>(a_s, a_d, mbf, NH);
    edge_max_kernel<<<((size_t)E * H + 255) / 256, 256, 0, stream>>>(ei, E, a_s, a_d, mbf, H);
    self_contrib_kernel<<<(NH + 255) / 256, 256, 0, stream>>>(h, a_s, a_d, mbf, den, out, N, H, C);
    edge_agg_kernel<<<(E + 7) / 8, 256, 0, stream>>>(ei, E, h, a_s, a_d, mbf, den, out, H, C);
  };

  // ---- Layer 1: 128 -> 8 heads x 32 (concat 256), ELU ----
  gemm_wmma_bf16<128, 256, 256, 2><<<(N + 31) / 32, 256, 0, stream>>>(featb, wf1, hbuf, N);
  edge_phase(hbuf, as1, ad1, 8, 32, obuf);
  finalize_bf16_kernel<<<((size_t)N * 256 + 255) / 256, 256, 0, stream>>>(
      obuf, den, b1, featb, N, 8, 32);   // feat2 overwrites xb (dead)

  // ---- Layer 2: 256 -> 1 x 32, ELU ----
  gemm_wmma_bf16<256, 32, 32, 4><<<(N + 63) / 64, 256, 0, stream>>>(featb, wf2, hbuf, N);
  edge_phase(hbuf, as2, ad2, 1, 32, sbuf);
  finalize_bf16_kernel<<<((size_t)N * 32 + 255) / 256, 256, 0, stream>>>(
      sbuf, den, b2, featb, N, 1, 32);   // feat3 overwrites feat2 (dead)

  // ---- Layer 3: 32 -> 1 x 40 (padded 48), logits f32 ----
  gemm_wmma_bf16<32, 48, 40, 4><<<(N + 63) / 64, 256, 0, stream>>>(featb, wf3, hbuf, N);
  edge_phase(hbuf, as3, ad3, 1, 40, obuf);
  finalize_f32_kernel<<<((size_t)N * 40 + 255) / 256, 256, 0, stream>>>(
      obuf, den, b3, obuf, N, 1, 40);

  log_softmax_kernel<<<(N + 7) / 8, 256, 0, stream>>>(obuf, (float*)d_out, N, 40);
}
